// RecSSM_88837103550848
// MI455X (gfx1250) — matrix operand order repaired and checked by reading, not compiled, not run on hardware
//
#include <hip/hip_runtime.h>
#include <hip/hip_bf16.h>
#include <math.h>

typedef _Float16 f16;
typedef __attribute__((ext_vector_type(16))) _Float16 v16h;
typedef __attribute__((ext_vector_type(8)))  float    v8f;

// ---- model constants ----
constexpr int BATCH = 16;
constexpr int LTOK  = 1024;            // 32x32 patches
constexpr int DM    = 128;             // d_model
constexpr int DI    = 256;             // d_inner
constexpr int DSTATE= 64;
constexpr int NH    = 8;
constexpr int HD    = 32;
constexpr int NL    = 4;
constexpr int DIP   = 2*DI + 2*DSTATE + NH;   // 648
constexpr int CD    = DI + 2*DSTATE;          // 384
constexpr int MTOK  = BATCH * LTOK;           // 16384

// fp16 weight copies in workspace
constexpr size_t W_PATCH = (size_t)128*768;
constexpr size_t W_IN    = (size_t)NL*DIP*DM;
constexpr size_t W_OUT   = (size_t)NL*DM*DI;
constexpr size_t W_A1    = (size_t)64*128;
constexpr size_t W_TOT   = W_PATCH + W_IN + W_OUT + W_A1;

// ------------------------------------------------------------------
// fp32 -> fp16 weight conversion (one pass over all weight tensors)
// ------------------------------------------------------------------
__global__ void cvt_weights(const float* __restrict__ pw, const float* __restrict__ inw,
                            const float* __restrict__ ow, const float* __restrict__ aw1,
                            f16* __restrict__ dst)
{
    size_t i = (size_t)blockIdx.x * blockDim.x + threadIdx.x;
    if (i < W_PATCH)                         dst[i] = (f16)pw[i];
    else if (i < W_PATCH+W_IN)               dst[i] = (f16)inw[i - W_PATCH];
    else if (i < W_PATCH+W_IN+W_OUT)         dst[i] = (f16)ow[i - W_PATCH - W_IN];
    else if (i < W_TOT)                      dst[i] = (f16)aw1[i - W_PATCH - W_IN - W_OUT];
}

// ------------------------------------------------------------------
// Generic WMMA GEMM: out[M,N] = A[M,K] * W[N,K]^T  (+bias, act, +=)
// block = 256 threads (8 waves); each wave owns one 16-wide N tile,
// block covers 128 N columns; grid = (M/16, ceil(N/128)).
// ------------------------------------------------------------------
template<int K, int ACT, int ACC>
__global__ void gemm_wmma(const float* __restrict__ A, const f16* __restrict__ W,
                          const float* __restrict__ bias, float* __restrict__ out, int N)
{
    const int wv   = threadIdx.x >> 5;
    const int lane = threadIdx.x & 31;
    const int n0   = blockIdx.y * 128 + wv * 16;
    if (n0 >= N) return;                       // tail waves idle (wave-uniform)
    const int m0   = blockIdx.x * 16;
    const int row  = lane & 15;
    const int hi   = lane >> 4;
    const int ncol = n0 + row;
    const f16*   wb = W + (size_t)((ncol < N) ? ncol : (N - 1)) * K;
    const float* ab = A + (size_t)(m0 + row) * K;

    v8f acc = {0.f,0.f,0.f,0.f,0.f,0.f,0.f,0.f};
#pragma unroll
    for (int k0 = 0; k0 < K; k0 += 32) {
        v16h a;
        const float* pa = ab + k0 + hi * 16;   // K layout: [hi*16 .. +15]
#pragma unroll
        for (int j = 0; j < 8; ++j) { a[j] = (f16)pa[j]; a[8+j] = (f16)pa[8+j]; }
        v16h b = *(const v16h*)(wb + k0 + hi * 16);   // 16 contiguous K per lane
        acc = __builtin_amdgcn_wmma_f32_16x16x32_f16(false, a, false, b,
                                                     (short)0, acc, false, false);
    }
    if (ncol < N) {
#pragma unroll
        for (int r = 0; r < 8; ++r) {
            int mm = m0 + (hi ? 8 + r : r);
            float v = acc[r];
            if (bias) v += bias[ncol];
            if (ACT == 1) v = tanhf(v);
            size_t o = (size_t)mm * N + ncol;
            if (ACC) out[o] += v; else out[o] = v;
        }
    }
}

// ------------------------------------------------------------------
// Patch-embed GEMM with gather: tok[m,128] = sum_k x_patch[m,k]*W[n,k] + b
// K flat index = c*256 + p*16 + q ; each 8-run is contiguous along q.
// ------------------------------------------------------------------
__global__ void patch_gemm(const float* __restrict__ x, const f16* __restrict__ W,
                           const float* __restrict__ bias, float* __restrict__ out)
{
    const int wv   = threadIdx.x >> 5;
    const int lane = threadIdx.x & 31;
    const int n0   = wv * 16;
    const int m0   = blockIdx.x * 16;
    const int row  = lane & 15;
    const int hi   = lane >> 4;
    const int m    = m0 + row;
    const int b    = m >> 10;
    const int idx  = m & 1023;
    const int hp   = idx >> 5;
    const int wpc  = idx & 31;
    const float* xb = x + ((size_t)b*3*512 + (size_t)hp*16)*512 + (size_t)wpc*16;
    const f16*   wb = W + (size_t)(n0 + row) * 768;

    v8f acc = {0.f,0.f,0.f,0.f,0.f,0.f,0.f,0.f};
#pragma unroll 4
    for (int k0 = 0; k0 < 768; k0 += 32) {
        const int ka = k0 + hi * 16;
        const int kb = ka + 8;
        const float* pa = xb + (size_t)(ka>>8)*262144 + (size_t)((ka>>4)&15)*512 + (ka&15);
        const float* pb = xb + (size_t)(kb>>8)*262144 + (size_t)((kb>>4)&15)*512 + (kb&15);
        v16h a;
#pragma unroll
        for (int j = 0; j < 8; ++j) { a[j] = (f16)pa[j]; a[8+j] = (f16)pb[j]; }
        v16h bf = *(const v16h*)(wb + k0 + hi * 16);
        acc = __builtin_amdgcn_wmma_f32_16x16x32_f16(false, a, false, bf,
                                                     (short)0, acc, false, false);
    }
    const int c = n0 + row;
#pragma unroll
    for (int r = 0; r < 8; ++r) {
        int mm = m0 + (hi ? 8 + r : r);
        out[(size_t)mm * DM + c] = acc[r] + bias[c];
    }
}

// ------------------------------------------------------------------
// LayerNorm over last dim 128, one token per block (128 threads)
// ------------------------------------------------------------------
__global__ void layernorm128(const float* __restrict__ in, const float* __restrict__ g,
                             const float* __restrict__ bb, float* __restrict__ out)
{
    __shared__ float s1[128], s2[128];
    const int mrow = blockIdx.x, t = threadIdx.x;
    float v = in[(size_t)mrow * DM + t];
    s1[t] = v; s2[t] = v * v;
    __syncthreads();
    for (int st = 64; st > 0; st >>= 1) {
        if (t < st) { s1[t] += s1[t+st]; s2[t] += s2[t+st]; }
        __syncthreads();
    }
    float mean = s1[0] * (1.f/128.f);
    float var  = s2[0] * (1.f/128.f) - mean*mean;
    out[(size_t)mrow * DM + t] = (v - mean) * rsqrtf(var + 1e-5f) * g[t] + bb[t];
}

// ------------------------------------------------------------------
// Depthwise causal conv (width 4) over xBC channels + SiLU
// ------------------------------------------------------------------
__global__ void conv_silu(const float* __restrict__ zx, const float* __restrict__ cw,
                          const float* __restrict__ cb, float* __restrict__ xc)
{
    size_t e = (size_t)blockIdx.x * 256 + threadIdx.x;
    if (e >= (size_t)MTOK * CD) return;
    int ch = (int)(e % CD);
    int l  = (int)((e / CD) & 1023);
    int b  = (int)(e / ((size_t)CD * LTOK));
    size_t base = (size_t)b * LTOK * DIP;
    float acc = cb[ch];
#pragma unroll
    for (int k = 0; k < 4; ++k) {
        int ll = l + k - 3;
        if (ll >= 0) acc += zx[base + (size_t)ll * DIP + DI + ch] * cw[ch*4 + k];
    }
    xc[e] = acc / (1.f + expf(-acc));      // SiLU
}

// ------------------------------------------------------------------
// Sequential SSM scan. One block per (batch, head); 256 threads:
// thread = (p, n-chunk of 8); state h[32,64] lives in registers.
// ------------------------------------------------------------------
__global__ void ssm_scan(const float* __restrict__ xc, const float* __restrict__ zx,
                         const float* __restrict__ dt_bias, const float* __restrict__ A_log,
                         const float* __restrict__ D_skip, float* __restrict__ y)
{
    const int b  = blockIdx.x >> 3;
    const int hh = blockIdx.x & 7;
    const int t  = threadIdx.x;
    const int p  = t >> 3;
    const int nc = t & 7;
    const int n0 = nc * 8;
    const float Ac  = -expf(A_log[hh]);
    const float dtb = dt_bias[hh];
    const float Dsk = D_skip[hh];
    float h[8];
#pragma unroll
    for (int j = 0; j < 8; ++j) h[j] = 0.f;
    const size_t xrow0 = (size_t)b * LTOK * CD;
    const size_t zrow0 = (size_t)b * LTOK * DIP;

    for (int l = 0; l < LTOK; ++l) {
        const size_t xr = xrow0 + (size_t)l * CD;
        float u  = zx[zrow0 + (size_t)l * DIP + (DIP - NH) + hh] + dtb;
        float dt = (u > 20.f) ? u : log1pf(expf(u));       // softplus
        float dA = expf(Ac * dt);
        float xt = xc[xr + hh * HD + p];
        const float4 B0 = *(const float4*)(xc + xr + DI + n0);
        const float4 B1 = *(const float4*)(xc + xr + DI + n0 + 4);
        const float4 C0 = *(const float4*)(xc + xr + DI + DSTATE + n0);
        const float4 C1 = *(const float4*)(xc + xr + DI + DSTATE + n0 + 4);
        if (l + 1 < LTOK) __builtin_prefetch(xc + xr + CD, 0, 0);
        const float cf = dt * xt;
        float acc;
        h[0] = h[0]*dA + cf*B0.x; acc  = h[0]*C0.x;
        h[1] = h[1]*dA + cf*B0.y; acc += h[1]*C0.y;
        h[2] = h[2]*dA + cf*B0.z; acc += h[2]*C0.z;
        h[3] = h[3]*dA + cf*B0.w; acc += h[3]*C0.w;
        h[4] = h[4]*dA + cf*B1.x; acc += h[4]*C1.x;
        h[5] = h[5]*dA + cf*B1.y; acc += h[5]*C1.y;
        h[6] = h[6]*dA + cf*B1.z; acc += h[6]*C1.z;
        h[7] = h[7]*dA + cf*B1.w; acc += h[7]*C1.w;
        // reduce across the 8 n-chunks sharing this p (lanes nc=0..7, wave32)
        acc += __shfl_xor(acc, 4);
        acc += __shfl_xor(acc, 2);
        acc += __shfl_xor(acc, 1);
        if (nc == 0)
            y[((size_t)b * LTOK + l) * DI + hh * HD + p] = acc + Dsk * xt;
    }
}

// ------------------------------------------------------------------
// y *= silu(z); RMSNorm over 256 with weight
// ------------------------------------------------------------------
__global__ void gate_rmsnorm(const float* __restrict__ y, const float* __restrict__ zx,
                             const float* __restrict__ nw, float* __restrict__ yn)
{
    __shared__ float s[256];
    const int mrow = blockIdx.x, t = threadIdx.x;
    float v = y[(size_t)mrow * DI + t];
    float z = zx[(size_t)mrow * DIP + t];
    v = v * (z / (1.f + expf(-z)));
    s[t] = v * v; __syncthreads();
    for (int st = 128; st > 0; st >>= 1) { if (t < st) s[t] += s[t+st]; __syncthreads(); }
    float ms = s[0] * (1.f/256.f);
    yn[(size_t)mrow * DI + t] = v * rsqrtf(ms + 1e-5f) * nw[t];
}

// ------------------------------------------------------------------
// scores[m] = h1[m,:64] . w2 + b2
// ------------------------------------------------------------------
__global__ void score_kernel(const float* __restrict__ h1, const float* __restrict__ w2,
                             const float* __restrict__ b2, float* __restrict__ sc)
{
    __shared__ float s[64];
    const int mrow = blockIdx.x, t = threadIdx.x;
    s[t] = h1[(size_t)mrow * 64 + t] * w2[t];
    __syncthreads();
    for (int st = 32; st > 0; st >>= 1) { if (t < st) s[t] += s[t+st]; __syncthreads(); }
    if (t == 0) sc[mrow] = s[0] + b2[0];
}

// ------------------------------------------------------------------
// Softmax over L per batch, weighted sum of tokens, L2-normalize.
// One block per batch, 256 threads.
// ------------------------------------------------------------------
__global__ void pool_kernel(const float* __restrict__ sc, const float* __restrict__ tok,
                            float* __restrict__ out)
{
    __shared__ float s[256];
    __shared__ float red[256];
    const int b = blockIdx.x, t = threadIdx.x;
    const float* sb = sc + (size_t)b * LTOK;
    float mx = -1e30f;
    for (int l = t; l < LTOK; l += 256) mx = fmaxf(mx, sb[l]);
    s[t] = mx; __syncthreads();
    for (int st = 128; st > 0; st >>= 1) { if (t < st) s[t] = fmaxf(s[t], s[t+st]); __syncthreads(); }
    mx = s[0]; __syncthreads();
    float se = 0.f;
    for (int l = t; l < LTOK; l += 256) se += expf(sb[l] - mx);
    s[t] = se; __syncthreads();
    for (int st = 128; st > 0; st >>= 1) { if (t < st) s[t] += s[t+st]; __syncthreads(); }
    se = s[0];
    const int d = t & 127, half = t >> 7;
    float acc = 0.f;
    const float* tb = tok + (size_t)b * LTOK * DM;
    for (int l = half * 512; l < half * 512 + 512; ++l)
        acc += expf(sb[l] - mx) * tb[(size_t)l * DM + d];
    red[t] = acc; __syncthreads();
    float pooled = 0.f;
    if (t < 128) pooled = (red[t] + red[t + 128]) / se;
    __syncthreads();
    s[t] = (t < 128) ? pooled * pooled : 0.f; __syncthreads();
    for (int st = 128; st > 0; st >>= 1) { if (t < st) s[t] += s[t+st]; __syncthreads(); }
    float nrm = fmaxf(sqrtf(s[0]), 1e-12f);
    if (t < 128) out[(size_t)b * DM + t] = pooled / nrm;
}

// ------------------------------------------------------------------
extern "C" void kernel_launch(void* const* d_in, const int* in_sizes, int n_in,
                              void* d_out, int out_size, void* d_ws, size_t ws_size,
                              hipStream_t stream)
{
    (void)in_sizes; (void)n_in; (void)out_size; (void)ws_size;
    const float* x        = (const float*)d_in[0];
    const float* patch_w  = (const float*)d_in[1];
    const float* patch_b  = (const float*)d_in[2];
    const float* pe_g     = (const float*)d_in[3];
    const float* pe_b     = (const float*)d_in[4];
    const float* in_w     = (const float*)d_in[5];
    const float* conv_w   = (const float*)d_in[6];
    const float* conv_b   = (const float*)d_in[7];
    const float* dt_bias  = (const float*)d_in[8];
    const float* A_log    = (const float*)d_in[9];
    const float* D_skip   = (const float*)d_in[10];
    const float* ssm_nw   = (const float*)d_in[11];
    const float* out_w    = (const float*)d_in[12];
    const float* ln_g     = (const float*)d_in[13];
    const float* ln_b     = (const float*)d_in[14];
    const float* fn_g     = (const float*)d_in[15];
    const float* fn_b     = (const float*)d_in[16];
    const float* attn_w1  = (const float*)d_in[17];
    const float* attn_b1  = (const float*)d_in[18];
    const float* attn_w2  = (const float*)d_in[19];
    const float* attn_b2  = (const float*)d_in[20];
    float* outp = (float*)d_out;

    char* ws = (char*)d_ws;
    size_t off = 0;
    auto take = [&](size_t bytes) -> char* {
        char* p = ws + off;
        off += (bytes + 255) & ~(size_t)255;
        return p;
    };
    f16*   wh     = (f16*)  take(W_TOT * sizeof(f16));
    float* tokbuf = (float*)take((size_t)MTOK * DM  * 4);
    float* lnbuf  = (float*)take((size_t)MTOK * DM  * 4);
    float* zx     = (float*)take((size_t)MTOK * DIP * 4);
    float* xc     = (float*)take((size_t)MTOK * CD  * 4);
    float* ybuf   = (float*)take((size_t)MTOK * DI  * 4);
    float* ynbuf  = (float*)take((size_t)MTOK * DI  * 4);
    float* h1buf  = (float*)take((size_t)MTOK * 64  * 4);
    float* scores = (float*)take((size_t)MTOK * 4);

    f16* pwh  = wh;
    f16* inwh = wh + W_PATCH;
    f16* owh  = inwh + W_IN;
    f16* a1h  = owh + W_OUT;

    // 1) convert weights to fp16 once per call
    cvt_weights<<<dim3((unsigned)((W_TOT + 255) / 256)), 256, 0, stream>>>(
        patch_w, in_w, out_w, attn_w1, wh);

    // 2) patch embedding + positional LN (in-place)
    patch_gemm<<<dim3(MTOK/16), 256, 0, stream>>>(x, pwh, patch_b, tokbuf);
    layernorm128<<<dim3(MTOK), 128, 0, stream>>>(tokbuf, pe_g, pe_b, tokbuf);

    // 3) Mamba-2 layers
    for (int l = 0; l < NL; ++l) {
        layernorm128<<<dim3(MTOK), 128, 0, stream>>>(tokbuf, ln_g + l*DM, ln_b + l*DM, lnbuf);
        gemm_wmma<DM, 0, 0><<<dim3(MTOK/16, (DIP + 127)/128), 256, 0, stream>>>(
            lnbuf, inwh + (size_t)l*DIP*DM, nullptr, zx, DIP);
        conv_silu<<<dim3((unsigned)(((size_t)MTOK*CD + 255) / 256)), 256, 0, stream>>>(
            zx, conv_w + (size_t)l*CD*4, conv_b + (size_t)l*CD, xc);
        ssm_scan<<<dim3(BATCH*NH), 256, 0, stream>>>(
            xc, zx, dt_bias + l*NH, A_log + l*NH, D_skip + l*NH, ybuf);
        gate_rmsnorm<<<dim3(MTOK), 256, 0, stream>>>(ybuf, zx, ssm_nw + (size_t)l*DI, ynbuf);
        gemm_wmma<DI, 0, 1><<<dim3(MTOK/16, 1), 256, 0, stream>>>(
            ynbuf, owh + (size_t)l*DM*DI, nullptr, tokbuf, DM);   // += residual
    }

    // 4) final LN + attention pooling
    layernorm128<<<dim3(MTOK), 128, 0, stream>>>(tokbuf, fn_g, fn_b, lnbuf);
    gemm_wmma<DM, 1, 0><<<dim3(MTOK/16, 1), 256, 0, stream>>>(
        lnbuf, a1h, attn_b1, h1buf, 64);                          // tanh fused
    score_kernel<<<dim3(MTOK), 64, 0, stream>>>(h1buf, attn_w2, attn_b2, scores);
    pool_kernel<<<dim3(BATCH), 256, 0, stream>>>(scores, lnbuf, outp);
}
